// BaselineAgent_2817498546572
// MI455X (gfx1250) — compile-verified, matrix operand order checked by reading, not run on hardware
//
#include <hip/hip_runtime.h>
#include <math.h>

typedef __attribute__((ext_vector_type(16))) _Float16 v16h;
typedef __attribute__((ext_vector_type(8)))  _Float16 v8h;
typedef __attribute__((ext_vector_type(4)))  _Float16 v4h;
typedef __attribute__((ext_vector_type(8)))  float    v8f;

#define WMMA16(A,B,C) __builtin_amdgcn_wmma_f32_16x16x32_f16(false,(A),false,(B),(short)0,(C),false,false)

static constexpr int T_STEPS = 64;
static constexpr int E_ENVS  = 32;
static constexpr int B_TOT   = 2048;   // T*E
static constexpr int HL      = 256;    // LSTM hidden
static constexpr int IND     = 549;    // LSTM input dim
static constexpr int KPAD    = 832;    // 549+256 padded to 26*32
static constexpr int XS      = 560;    // X row stride (f16 elems)
static constexpr int ASTR    = 840;    // LDS A row stride (f16 elems)
static constexpr int IMS     = 88;     // padded CHW image row stride (f16 elems)

// ---------------------------------------------------------------------------
// Fragment packing per CDNA5 WMMA 16-bit A/B VGPR layout:
//   lanes 0-15 : K = {kb+0..7, kb+16..23}   lanes 16-31: K = {kb+8..15, kb+24..31}
// caller passes p = &row[kb + laneHi*8]; element e<8 -> p[e], e>=8 -> p[16+e-8].
__device__ __forceinline__ v16h pack_frag(const _Float16* p) {
  v8h u = *(const v8h*)p;
  v8h w = *(const v8h*)(p + 16);
  v16h a;
#pragma unroll
  for (int e = 0; e < 8; e++) { a[e] = u[e]; a[e + 8] = w[e]; }
  return a;
}
// conv1 fragment from padded CHW image: two rows (kh, kh+2), 8 cols each, 8B loads
__device__ __forceinline__ v16h pack_frag_c1(const _Float16* p0) {
  const _Float16* p1 = p0 + 2 * IMS;
  v4h u0 = *(const v4h*)p0;
  v4h u1 = *(const v4h*)(p0 + 4);
  v4h u2 = *(const v4h*)p1;
  v4h u3 = *(const v4h*)(p1 + 4);
  v16h a;
#pragma unroll
  for (int e = 0; e < 4; e++) {
    a[e] = u0[e]; a[e + 4] = u1[e]; a[e + 8] = u2[e]; a[e + 12] = u3[e];
  }
  return a;
}

__device__ __forceinline__ float sigmoidf_(float x) { return 1.0f / (1.0f + __expf(-x)); }

// ---------------------------------------------------------------------------
// prep kernels
// image NHWC f32 -> padded CHW f16: imgC[b][c][row][0..83] stride 88, /255
__global__ void k_prep_img(const float* img, _Float16* imgC, int n) {
  int i = blockIdx.x * blockDim.x + threadIdx.x;
  if (i >= n) return;                      // n = B*3*84*84
  int b = i / (3 * 84 * 84), r0 = i % (3 * 84 * 84);
  int c = r0 / (84 * 84), rc = r0 % (84 * 84);
  int row = rc / 84, col = rc % 84;
  imgC[((b * 3 + c) * 84 + row) * IMS + col] =
      (_Float16)(img[((b * 84 + row) * 84 + col) * 3 + c] * (1.0f / 255.0f));
}

// conv1 weights [32,3,8,8] OIHW -> w1t[n][k], k = c*64 + kh*8 + kw
__global__ void k_prep_w1(const float* w, _Float16* wt) {
  int i = blockIdx.x * blockDim.x + threadIdx.x;
  if (i >= 32 * 192) return;
  int n = i / 192, k = i % 192;
  int c = k >> 6, kh = (k & 63) >> 3, kw = k & 7;
  wt[i] = (_Float16)w[((n * 3 + c) * 8 + kh) * 8 + kw];
}
// conv2 weights [64,32,4,4] -> w2t[n][k], k = (kh*4+kw)*32 + c
__global__ void k_prep_w2(const float* w, _Float16* wt) {
  int i = blockIdx.x * blockDim.x + threadIdx.x;
  if (i >= 64 * 512) return;
  int n = i / 512, k = i % 512;
  int pix = k >> 5, c = k & 31, kh = pix >> 2, kw = pix & 3;
  wt[i] = (_Float16)w[((n * 32 + c) * 4 + kh) * 4 + kw];
}
// conv3 weights [64,64,3,3] -> w3t[n][k], k = (kh*3+kw)*64 + c
__global__ void k_prep_w3(const float* w, _Float16* wt) {
  int i = blockIdx.x * blockDim.x + threadIdx.x;
  if (i >= 64 * 576) return;
  int n = i / 576, k = i % 576;
  int pix = k >> 6, c = k & 63, kh = pix / 3, kw = pix % 3;
  wt[i] = (_Float16)w[((n * 64 + c) * 3 + kh) * 3 + kw];
}
// fc_w [512,3136] with col = c*49+s  ->  fcwt[f][k], k = s*64+c (NHWC order)
__global__ void k_prep_fc(const float* w, _Float16* wt) {
  int i = blockIdx.x * blockDim.x + threadIdx.x;
  if (i >= 512 * 3136) return;
  int f = i / 3136, k = i % 3136;
  int s = k >> 6, c = k & 63;
  wt[i] = (_Float16)w[f * 3136 + c * 49 + s];
}
// combined LSTM weights: Wg[1024][832] = [W_ih | W_hh | 0pad]
__global__ void k_prep_wg(const float* wih, const float* whh, _Float16* wg) {
  int i = blockIdx.x * blockDim.x + threadIdx.x;
  if (i >= 1024 * KPAD) return;
  int r = i / KPAD, k = i % KPAD;
  float v = 0.0f;
  if (k < IND)            v = wih[r * IND + k];
  else if (k < IND + HL)  v = whh[r * HL + (k - IND)];
  wg[i] = (_Float16)v;
}
__global__ void k_prep_bias(const float* bih, const float* bhh, float* bg) {
  int i = blockIdx.x * blockDim.x + threadIdx.x;
  if (i < 1024) bg[i] = bih[i] + bhh[i];
}
// one-hot block of X: cols 512..548
__global__ void k_onehot(const int* pos, const int* lact, _Float16* X) {
  int i = blockIdx.x * blockDim.x + threadIdx.x;
  if (i >= B_TOT * 48) return;
  int b = i / 48, c = i % 48;
  float v = 0.0f;
  if (c < 16)      v = (pos[b * 2 + 0] == c)      ? 1.0f : 0.0f;
  else if (c < 32) v = (pos[b * 2 + 1] == c - 16) ? 1.0f : 0.0f;
  else if (c < 37) v = (lact[b]        == c - 32) ? 1.0f : 0.0f;
  X[b * XS + 512 + c] = (_Float16)v;
}

// ---------------------------------------------------------------------------
// Conv1 implicit GEMM: M = B*400, N = 32, K = 192 (k = c*64+kh*8+kw, CHW input)
// one wave = 16 rows x full N=32 (2 accumulators), A loaded once per chunk.
__global__ void __launch_bounds__(256) k_conv1(const _Float16* imgC, const _Float16* wt,
                                               const float* bias, _Float16* out) {
  int wid  = (blockIdx.x * blockDim.x + threadIdx.x) >> 5;   // = mtile
  int lane = threadIdx.x & 31, hi = lane >> 4, lo = lane & 15;
  int m = wid * 16 + lo;
  int b = m / 400, p = m % 400, oh = p / 20, ow = p % 20;
  const _Float16* abase = imgC + (size_t)b * 3 * 84 * IMS + (oh * 4) * IMS + ow * 4;
  v8f acc0 = {}, acc1 = {};
#pragma unroll
  for (int kb = 0; kb < 192; kb += 32) {
    int c = kb >> 6, khb = (kb & 63) >> 3;
    v16h a  = pack_frag_c1(abase + (c * 84 + khb + hi) * IMS);
    v16h b0 = pack_frag(wt + lo * 192        + kb + hi * 8);
    v16h b1 = pack_frag(wt + (16 + lo) * 192 + kb + hi * 8);
    acc0 = WMMA16(a, b0, acc0);
    acc1 = WMMA16(a, b1, acc1);
  }
  float bs0 = bias[lo], bs1 = bias[16 + lo];
#pragma unroll
  for (int r = 0; r < 8; r++) {
    int mm = wid * 16 + hi * 8 + r;
    float v0 = acc0[r] + bs0; v0 = v0 > 0.0f ? v0 : 0.0f;
    float v1 = acc1[r] + bs1; v1 = v1 > 0.0f ? v1 : 0.0f;
    out[mm * 32 + lo]      = (_Float16)v0;
    out[mm * 32 + 16 + lo] = (_Float16)v1;
  }
}

// Conv2 implicit GEMM: M = B*81, N = 64, K = 512; one wave = 16 rows x full N (4 acc)
__global__ void __launch_bounds__(256) k_conv2(const _Float16* act1, const _Float16* wt,
                                               const float* bias, _Float16* out) {
  int wid  = (blockIdx.x * blockDim.x + threadIdx.x) >> 5;   // = mtile
  int lane = threadIdx.x & 31, hi = lane >> 4, lo = lane & 15;
  int m = wid * 16 + lo;
  int b = m / 81, p = m % 81, oh = p / 9, ow = p % 9;
  v8f acc0 = {}, acc1 = {}, acc2 = {}, acc3 = {};
  for (int kb = 0; kb < 512; kb += 32) {
    int pix = kb >> 5, kh = pix >> 2, kw = pix & 3;
    v16h a  = pack_frag(act1 + ((b * 20 + oh * 2 + kh) * 20 + (ow * 2 + kw)) * 32 + hi * 8);
    v16h b0 = pack_frag(wt + (lo)      * 512 + kb + hi * 8);
    v16h b1 = pack_frag(wt + (16 + lo) * 512 + kb + hi * 8);
    v16h b2 = pack_frag(wt + (32 + lo) * 512 + kb + hi * 8);
    v16h b3 = pack_frag(wt + (48 + lo) * 512 + kb + hi * 8);
    acc0 = WMMA16(a, b0, acc0);
    acc1 = WMMA16(a, b1, acc1);
    acc2 = WMMA16(a, b2, acc2);
    acc3 = WMMA16(a, b3, acc3);
  }
  float bs0 = bias[lo], bs1 = bias[16 + lo], bs2 = bias[32 + lo], bs3 = bias[48 + lo];
#pragma unroll
  for (int r = 0; r < 8; r++) {
    int mm = wid * 16 + hi * 8 + r;
    float v0 = acc0[r] + bs0; v0 = v0 > 0.0f ? v0 : 0.0f;
    float v1 = acc1[r] + bs1; v1 = v1 > 0.0f ? v1 : 0.0f;
    float v2 = acc2[r] + bs2; v2 = v2 > 0.0f ? v2 : 0.0f;
    float v3 = acc3[r] + bs3; v3 = v3 > 0.0f ? v3 : 0.0f;
    out[mm * 64 + lo]      = (_Float16)v0;
    out[mm * 64 + 16 + lo] = (_Float16)v1;
    out[mm * 64 + 32 + lo] = (_Float16)v2;
    out[mm * 64 + 48 + lo] = (_Float16)v3;
  }
}

// Conv3 implicit GEMM: M = B*49, N = 64, K = 576; one wave = 16 rows x full N (4 acc)
__global__ void __launch_bounds__(256) k_conv3(const _Float16* act2, const _Float16* wt,
                                               const float* bias, _Float16* out) {
  int wid  = (blockIdx.x * blockDim.x + threadIdx.x) >> 5;   // = mtile
  int lane = threadIdx.x & 31, hi = lane >> 4, lo = lane & 15;
  int m = wid * 16 + lo;
  int b = m / 49, p = m % 49, oh = p / 7, ow = p % 7;
  v8f acc0 = {}, acc1 = {}, acc2 = {}, acc3 = {};
  for (int kb = 0; kb < 576; kb += 32) {
    int pix = kb >> 6, cb = (kb & 63) + hi * 8;
    int kh = pix / 3, kw = pix % 3;
    v16h a  = pack_frag(act2 + ((b * 9 + oh + kh) * 9 + (ow + kw)) * 64 + cb);
    v16h b0 = pack_frag(wt + (lo)      * 576 + kb + hi * 8);
    v16h b1 = pack_frag(wt + (16 + lo) * 576 + kb + hi * 8);
    v16h b2 = pack_frag(wt + (32 + lo) * 576 + kb + hi * 8);
    v16h b3 = pack_frag(wt + (48 + lo) * 576 + kb + hi * 8);
    acc0 = WMMA16(a, b0, acc0);
    acc1 = WMMA16(a, b1, acc1);
    acc2 = WMMA16(a, b2, acc2);
    acc3 = WMMA16(a, b3, acc3);
  }
  float bs0 = bias[lo], bs1 = bias[16 + lo], bs2 = bias[32 + lo], bs3 = bias[48 + lo];
#pragma unroll
  for (int r = 0; r < 8; r++) {
    int mm = wid * 16 + hi * 8 + r;
    float v0 = acc0[r] + bs0; v0 = v0 > 0.0f ? v0 : 0.0f;
    float v1 = acc1[r] + bs1; v1 = v1 > 0.0f ? v1 : 0.0f;
    float v2 = acc2[r] + bs2; v2 = v2 > 0.0f ? v2 : 0.0f;
    float v3 = acc3[r] + bs3; v3 = v3 > 0.0f ? v3 : 0.0f;
    out[mm * 64 + lo]      = (_Float16)v0;
    out[mm * 64 + 16 + lo] = (_Float16)v1;
    out[mm * 64 + 32 + lo] = (_Float16)v2;
    out[mm * 64 + 48 + lo] = (_Float16)v3;
  }
}

// FC GEMM: M = 2048, N = 512, K = 3136; one wave = 16 rows x 64 cols (4 acc);
// relu output written into X[b][0..511]
__global__ void __launch_bounds__(256) k_fc(const _Float16* act3, const _Float16* wt,
                                            const float* bias, _Float16* X) {
  int wid  = (blockIdx.x * blockDim.x + threadIdx.x) >> 5;
  int lane = threadIdx.x & 31, hi = lane >> 4, lo = lane & 15;
  int ng = wid & 7, mt = wid >> 3;           // 8 n-groups of 64
  int m = mt * 16 + lo;
  int nb = ng * 64;
  v8f acc0 = {}, acc1 = {}, acc2 = {}, acc3 = {};
  for (int kb = 0; kb < 3136; kb += 32) {
    v16h a  = pack_frag(act3 + (size_t)m * 3136 + kb + hi * 8);
    v16h b0 = pack_frag(wt + (size_t)(nb + lo)      * 3136 + kb + hi * 8);
    v16h b1 = pack_frag(wt + (size_t)(nb + 16 + lo) * 3136 + kb + hi * 8);
    v16h b2 = pack_frag(wt + (size_t)(nb + 32 + lo) * 3136 + kb + hi * 8);
    v16h b3 = pack_frag(wt + (size_t)(nb + 48 + lo) * 3136 + kb + hi * 8);
    acc0 = WMMA16(a, b0, acc0);
    acc1 = WMMA16(a, b1, acc1);
    acc2 = WMMA16(a, b2, acc2);
    acc3 = WMMA16(a, b3, acc3);
  }
  float bs0 = bias[nb + lo], bs1 = bias[nb + 16 + lo];
  float bs2 = bias[nb + 32 + lo], bs3 = bias[nb + 48 + lo];
#pragma unroll
  for (int r = 0; r < 8; r++) {
    int mm = mt * 16 + hi * 8 + r;
    float v0 = acc0[r] + bs0; v0 = v0 > 0.0f ? v0 : 0.0f;
    float v1 = acc1[r] + bs1; v1 = v1 > 0.0f ? v1 : 0.0f;
    float v2 = acc2[r] + bs2; v2 = v2 > 0.0f ? v2 : 0.0f;
    float v3 = acc3[r] + bs3; v3 = v3 > 0.0f ? v3 : 0.0f;
    X[mm * XS + nb + lo]      = (_Float16)v0;
    X[mm * XS + nb + 16 + lo] = (_Float16)v1;
    X[mm * XS + nb + 32 + lo] = (_Float16)v2;
    X[mm * XS + nb + 48 + lo] = (_Float16)v3;
  }
}

// ---------------------------------------------------------------------------
// Masked LSTM: one persistent 1024-thread workgroup (32 waves).
// LDS A[32][ASTR] holds concat[x_t | h | 0pad]; c kept in registers (each wave
// exclusively owns rows mt*16..+15 x gate-cols cg*16..+15, all 4 gates).
__global__ void __launch_bounds__(1024) k_lstm(const _Float16* X, const _Float16* Wg,
                                               const float* biasg, const int* done,
                                               const float* h0, const float* c0,
                                               float* hs, float* out) {
  __shared__ _Float16 A[E_ENVS * ASTR];
  int tid = threadIdx.x;
  int w = tid >> 5, lane = tid & 31, hi = lane >> 4, lo = lane & 15;
  int mt = w >> 4, cg = w & 15;
  int ncol = cg * 16 + lo;

  // init h region of A and zero tail padding
  for (int i = tid; i < E_ENVS * HL; i += 1024) {
    int e = i >> 8, k = i & 255;
    A[e * ASTR + IND + k] = (_Float16)h0[i];
  }
  for (int i = tid; i < E_ENVS * (ASTR - (IND + HL)); i += 1024) {
    int span = ASTR - (IND + HL);
    int e = i / span, k = i % span;
    A[e * ASTR + IND + HL + k] = (_Float16)0.0f;
  }
  v8f creg;
#pragma unroll
  for (int r = 0; r < 8; r++) {
    int env = mt * 16 + hi * 8 + r;
    creg[r] = c0[env * HL + ncol];
  }
  float bi = biasg[0 * HL + ncol], bf = biasg[1 * HL + ncol];
  float bg = biasg[2 * HL + ncol], bo = biasg[3 * HL + ncol];
  __syncthreads();

  for (int t = 0; t < T_STEPS; t++) {
    // done-mask h in LDS, load x_t, done-mask c registers
    for (int i = tid; i < E_ENVS * HL; i += 1024) {
      int e = i >> 8, k = i & 255;
      if (done[t * E_ENVS + e]) A[e * ASTR + IND + k] = (_Float16)0.0f;
    }
    for (int i = tid; i < E_ENVS * IND; i += 1024) {
      int e = i / IND, k = i - e * IND;
      A[e * ASTR + k] = X[(t * E_ENVS + e) * XS + k];
    }
#pragma unroll
    for (int r = 0; r < 8; r++) {
      int env = mt * 16 + hi * 8 + r;
      if (done[t * E_ENVS + env]) creg[r] = 0.0f;
    }
    __syncthreads();

    // gates = [x|h] @ Wg^T + bias : 4 WMMA chains (i,f,g,o), K = 26 x 32
    v8f ai, af, ag, ao;
#pragma unroll
    for (int r = 0; r < 8; r++) { ai[r] = bi; af[r] = bf; ag[r] = bg; ao[r] = bo; }
    for (int kb = 0; kb < KPAD; kb += 32) {
      v16h a  = pack_frag(&A[(mt * 16 + lo) * ASTR + kb + hi * 8]);
      v16h b0 = pack_frag(Wg + (0 * HL + ncol) * KPAD + kb + hi * 8);
      v16h b1 = pack_frag(Wg + (1 * HL + ncol) * KPAD + kb + hi * 8);
      v16h b2 = pack_frag(Wg + (2 * HL + ncol) * KPAD + kb + hi * 8);
      v16h b3 = pack_frag(Wg + (3 * HL + ncol) * KPAD + kb + hi * 8);
      ai = WMMA16(a, b0, ai);
      af = WMMA16(a, b1, af);
      ag = WMMA16(a, b2, ag);
      ao = WMMA16(a, b3, ao);
    }
    __syncthreads();  // all LDS reads of h done before overwriting

    float hv[8];
#pragma unroll
    for (int r = 0; r < 8; r++) {
      float iv = sigmoidf_(ai[r]), fv = sigmoidf_(af[r]);
      float gv = tanhf(ag[r]),     ov = sigmoidf_(ao[r]);
      float cv = fv * creg[r] + iv * gv;
      creg[r] = cv;
      hv[r]   = ov * tanhf(cv);
    }
#pragma unroll
    for (int r = 0; r < 8; r++) {
      int env = mt * 16 + hi * 8 + r;
      A[env * ASTR + IND + ncol] = (_Float16)hv[r];
      hs[(t * E_ENVS + env) * HL + ncol] = hv[r];
      if (t == T_STEPS - 1) {
        out[12288 + env * HL + ncol] = hv[r];   // h_n
        out[20480 + env * HL + ncol] = creg[r]; // c_n
      }
    }
    __syncthreads();
  }
}

// Policy/value heads: 2048x6 dots of length 256 (tiny, VALU)
__global__ void k_heads(const float* hs, const float* pw, const float* pb,
                        const float* vw, const float* vb, float* out) {
  int i = blockIdx.x * blockDim.x + threadIdx.x;
  if (i >= B_TOT * 6) return;
  int b = i / 6, j = i % 6;
  const float* h    = hs + b * HL;
  const float* wrow = (j < 5) ? pw + j * HL : vw;
  float s = (j < 5) ? pb[j] : vb[0];
#pragma unroll 4
  for (int k = 0; k < HL; k++) s += h[k] * wrow[k];
  if (j < 5) out[b * 5 + j] = s;
  else       out[10240 + b] = s;
}

// ---------------------------------------------------------------------------
extern "C" void kernel_launch(void* const* d_in, const int* in_sizes, int n_in,
                              void* d_out, int out_size, void* d_ws, size_t ws_size,
                              hipStream_t stream) {
  (void)in_sizes; (void)n_in; (void)out_size; (void)ws_size;
  const float* img  = (const float*)d_in[0];
  const int*   pos  = (const int*)d_in[1];
  const int*   lact = (const int*)d_in[2];
  const int*   done = (const int*)d_in[3];
  const float* h0   = (const float*)d_in[4];
  const float* c0   = (const float*)d_in[5];
  const float* w1   = (const float*)d_in[6];
  const float* b1   = (const float*)d_in[7];
  const float* w2   = (const float*)d_in[8];
  const float* b2   = (const float*)d_in[9];
  const float* w3   = (const float*)d_in[10];
  const float* b3   = (const float*)d_in[11];
  const float* fcw  = (const float*)d_in[12];
  const float* fcb  = (const float*)d_in[13];
  const float* wih  = (const float*)d_in[14];
  const float* whh  = (const float*)d_in[15];
  const float* bih  = (const float*)d_in[16];
  const float* bhh  = (const float*)d_in[17];
  const float* pw   = (const float*)d_in[18];
  const float* pb   = (const float*)d_in[19];
  const float* vw   = (const float*)d_in[20];
  const float* vb   = (const float*)d_in[21];
  float* out = (float*)d_out;
  char*  ws  = (char*)d_ws;

  // workspace layout (bytes); imgC region is reused for act2/act3/X/hs after conv1
  const size_t o_img  = 0;                     // 90,832,896 B (B*3*84*88 f16)
  const size_t o_act1 = 90832896;              // 52,428,800 B
  const size_t o_w1t  = o_act1 + 52428800;     // 12,288
  const size_t o_w2t  = o_w1t + 12288;         // 65,536
  const size_t o_w3t  = o_w2t + 65536;         // 73,728
  const size_t o_fcwt = o_w3t + 73728;         // 3,211,264
  const size_t o_wg   = o_fcwt + 3211264;      // 1,703,936
  const size_t o_bg   = o_wg + 1703936;        // 4,096
  // aliases inside imgC region (valid after conv1 completes):
  const size_t o_act2 = 0;                     // 21,233,664
  const size_t o_act3 = 21233664;              // 12,845,056
  const size_t o_X    = 34078720;              // 2,293,760
  const size_t o_hs   = 36372480;              // 2,097,152

  _Float16* imgC = (_Float16*)(ws + o_img);
  _Float16* act1 = (_Float16*)(ws + o_act1);
  _Float16* act2 = (_Float16*)(ws + o_act2);
  _Float16* act3 = (_Float16*)(ws + o_act3);
  _Float16* Xm   = (_Float16*)(ws + o_X);
  _Float16* w1t  = (_Float16*)(ws + o_w1t);
  _Float16* w2t  = (_Float16*)(ws + o_w2t);
  _Float16* w3t  = (_Float16*)(ws + o_w3t);
  _Float16* fcwt = (_Float16*)(ws + o_fcwt);
  _Float16* wg   = (_Float16*)(ws + o_wg);
  float*    bg   = (float*)(ws + o_bg);
  float*    hsb  = (float*)(ws + o_hs);

  const int TB = 256;
  // weight / input prep
  int nImg = B_TOT * 3 * 84 * 84;
  k_prep_img <<<(nImg + TB - 1) / TB, TB, 0, stream>>>(img, imgC, nImg);
  k_prep_w1  <<<(32 * 192   + TB - 1) / TB, TB, 0, stream>>>(w1, w1t);
  k_prep_w2  <<<(64 * 512   + TB - 1) / TB, TB, 0, stream>>>(w2, w2t);
  k_prep_w3  <<<(64 * 576   + TB - 1) / TB, TB, 0, stream>>>(w3, w3t);
  k_prep_fc  <<<(512 * 3136 + TB - 1) / TB, TB, 0, stream>>>(fcw, fcwt);
  k_prep_wg  <<<(1024 * KPAD + TB - 1) / TB, TB, 0, stream>>>(wih, whh, wg);
  k_prep_bias<<<(1024 + TB - 1) / TB, TB, 0, stream>>>(bih, bhh, bg);

  // convs: one wave per 16-row M-tile covering the full N (multi-acc); FC: 16 rows x 64 cols/wave;
  // exact grids, 8 waves / 256-thread block
  k_conv1<<<6400, TB, 0, stream>>>(imgC, w1t, b1, act1);   // 51200 mtiles
  k_conv2<<<1296, TB, 0, stream>>>(act1, w2t, b2, act2);   // 10368 mtiles
  k_conv3<<< 784, TB, 0, stream>>>(act2, w3t, b3, act3);   //  6272 mtiles
  k_fc   <<< 128, TB, 0, stream>>>(act3, fcwt, fcb, Xm);   //   128 mt * 8 ngroups

  // one-hot features into X[., 512..548]
  k_onehot<<<(B_TOT * 48 + TB - 1) / TB, TB, 0, stream>>>(pos, lact, Xm);

  // sequential masked LSTM (single workgroup, 32 waves)
  k_lstm<<<1, 1024, 0, stream>>>(Xm, wg, bg, done, h0, c0, hsb, out);

  // heads
  k_heads<<<(B_TOT * 6 + TB - 1) / TB, TB, 0, stream>>>(hsb, pw, pb, vw, vb, out);
}
